// ConvertedBlockGRU_19490561589816
// MI455X (gfx1250) — matrix operand order repaired and checked by reading, not compiled, not run on hardware
//
#include <hip/hip_runtime.h>

// ---------------------------------------------------------------------------
// Block GRU (2 layers, NB=8 blocks, CH=CIN=256, B=64, T=256) for gfx1250.
//
// Phase A: UX = bf16-WMMA GEMM of all timesteps' input projections (+ b_ih),
//          M=131072, K=256, N=768, stored [seq][t][768] (seq = b*8+nb).
//          Weight slice (48KB) staged to LDS via TDM tensor_load_to_lds.
// Phase B: sequential recurrence. 32 WGs x 16 waves; each WG owns 16
//          sequences; each wave owns a 16-wide hidden slice and keeps its
//          3x8 = 24 W_hh bf16 B-fragments (192 VGPRs) resident for all 256
//          steps. h state double-buffered in LDS (2 x 8KB), 1 barrier/step.
// ---------------------------------------------------------------------------

typedef __attribute__((ext_vector_type(16))) __bf16 v16bf;
typedef __attribute__((ext_vector_type(8)))  float v8f;
typedef __attribute__((ext_vector_type(4)))  unsigned int uint4n; // trivial 16B vector
typedef __attribute__((ext_vector_type(4)))  unsigned int u32x4;
typedef __attribute__((ext_vector_type(8)))  int i32x8;
typedef __attribute__((ext_vector_type(4)))  int i32x4;

union BFrag { uint4n q[2]; v16bf v; };

#define NBLK   8
#define CH     256
#define TT     256
#define BB     64
#define NSEQ   (BB * NBLK)              // 512 sequences
#define MROWS  (BB * TT * NBLK)         // 131072 GEMM rows
#define NGATE  (3 * CH)                 // 768

__device__ __forceinline__ unsigned short f2bf(float f) {
    unsigned u = __float_as_uint(f);
    u += 0x7FFFu + ((u >> 16) & 1u);    // round-to-nearest-even
    return (unsigned short)(u >> 16);
}
__device__ __forceinline__ float bf2f(unsigned short s) {
    return __uint_as_float(((unsigned)s) << 16);
}
__device__ __forceinline__ float fsig(float x) { return 1.0f / (1.0f + __expf(-x)); }
__device__ __forceinline__ float ftanh(float x) {
    float e = __expf(-2.0f * x);
    return (1.0f - e) / (1.0f + e);
}
__device__ __forceinline__ v8f wmma_bf16(const BFrag& a, const BFrag& b, v8f c) {
    return __builtin_amdgcn_wmma_f32_16x16x32_bf16(false, a.v, false, b.v,
                                                   (short)0, c, false, false);
}

// TDM 1-D block copy global -> LDS. `bytes` must be a multiple of 8.
// D# packing per CDNA5 ISA ch.8: group0 {count=1, lds_addr, global_addr,
// type=2}; group1 {data_size=8B, tensor_dim0=tile_dim0=bytes/8, dim1=1,
// stride0=bytes/8}; remaining groups zero (<=2D tensor). Issue from ONE wave.
// This toolchain declares the 6-arg builtin form.
__device__ __forceinline__ void tdm_load_1d(const void* gsrc, void* ldst,
                                            unsigned bytes) {
    unsigned long long ga = (unsigned long long)gsrc;
    unsigned lds = (unsigned)(unsigned long long)ldst;  // low 32 bits = LDS offset
    unsigned n8 = bytes >> 3;

    u32x4 g0;
    g0[0] = 1u;                                         // count=1
    g0[1] = lds;                                        // lds_addr
    g0[2] = (unsigned)(ga & 0xFFFFFFFFu);               // global_addr[31:0]
    g0[3] = (unsigned)((ga >> 32) & 0x1FFFFFFu) | (2u << 30); // [56:32] | type=2

    i32x8 g1;
    g1[0] = (int)(3u << 16);                 // workgroup_mask=0, data_size=3 (8B)
    g1[1] = (int)((n8 & 0xFFFFu) << 16);     // tensor_dim0[15:0] at bits 63:48
    g1[2] = (int)(((n8 >> 16) & 0xFFFFu) | (1u << 16)); // dim0[31:16], tensor_dim1=1
    g1[3] = (int)((n8 & 0xFFFFu) << 16);     // tile_dim0 at bits 127:112
    g1[4] = 1;                               // tile_dim1=1, tile_dim2=0
    g1[5] = (int)n8;                         // tensor_dim0_stride[31:0]
    g1[6] = 0;                               // stride hi, tensor_dim1_stride lo
    g1[7] = 0;

    i32x4 z4 = {0, 0, 0, 0};
    i32x8 z8 = {0, 0, 0, 0, 0, 0, 0, 0};
    __builtin_amdgcn_tensor_load_to_lds(g0, g1, z4, z4, z8, 0);
}

// ---------------------------------------------------------------------------
__global__ void f32_to_bf16_kernel(const float* __restrict__ in,
                                   unsigned short* __restrict__ out, long n) {
    long i = (long)blockIdx.x * blockDim.x + threadIdx.x;
    long stride = (long)gridDim.x * blockDim.x;
    for (; i < n; i += stride) out[i] = f2bf(in[i]);
}

// ---------------------------------------------------------------------------
// WMMA GEMM: UX[seq*T + t][n] = sum_k A[row][k] * W[n][k] + bias[n]
// Grid: (1024, 8). Block 256 (8 waves). WG tile: M=128 (16 rows/wave), N=96.
__global__ __launch_bounds__(256) void wmma_gemm_ux(
    const unsigned short* __restrict__ A,
    const unsigned short* __restrict__ W,     // 768 x 256 bf16 row-major
    const float* __restrict__ bias,           // 768
    float* __restrict__ UX,                   // [seq][t][768]
    int a_perm)
{
    const int tid  = threadIdx.x;
    const int wave = tid >> 5, lane = tid & 31;
    const int h8 = lane >> 4, r = lane & 15;
    const int n0 = blockIdx.y * 96;
    const long R0 = (long)blockIdx.x * 128 + wave * 16;

    __shared__ __align__(16) unsigned short wlds[96 * 256]; // 48 KB W slice

    // Stage weight slice (rows n0..n0+95 are one contiguous 48KB block)
    // via the Tensor Data Mover: one wave issues, waits TENSORcnt, barrier.
    if (wave == 0) {
        tdm_load_1d(W + (size_t)n0 * 256, wlds, 96 * 256 * 2);
        __builtin_amdgcn_s_wait_tensorcnt(0);
    }
    __syncthreads();

    const v8f vzero = {0.f, 0.f, 0.f, 0.f, 0.f, 0.f, 0.f, 0.f};
    v8f acc[6];
    #pragma unroll
    for (int i = 0; i < 6; ++i) acc[i] = vzero;

    // A-fragment row for this lane
    size_t abase;
    {
        long row = R0 + r;
        if (a_perm) {
            long nb = row & 7, bt = row >> 3;
            long t = bt & (TT - 1), b = bt >> 8;
            abase = (size_t)((b * NBLK + nb) * TT + t) * CH;
        } else {
            abase = (size_t)row * CH;
        }
    }
    const unsigned short* ap = A + abase;

    #pragma unroll
    for (int kt = 0; kt < 8; ++kt) {
        BFrag af;
        const int ke = kt * 32 + 8 * h8;          // A: elems 0..7 at ke, 8..15 at ke+16
        af.q[0] = *(const uint4n*)(ap + ke);
        af.q[1] = *(const uint4n*)(ap + ke + 16);
        // Gather all 6 B-fragments first so the ds loads clause together and
        // the 6 WMMAs issue back-to-back behind one s_wait_dscnt.
        BFrag bf[6];
        #pragma unroll
        for (int nt = 0; nt < 6; ++nt) {
            const unsigned short* wp = wlds + (nt * 16 + r) * 256 + kt * 32 + 8 * h8;
            bf[nt].q[0] = *(const uint4n*)(wp);
            bf[nt].q[1] = *(const uint4n*)(wp + 16);
        }
        #pragma unroll
        for (int nt = 0; nt < 6; ++nt)
            acc[nt] = wmma_bf16(af, bf[nt], acc[nt]);
    }

    // Epilogue: add bias, store seq-major UX.
    #pragma unroll
    for (int nt = 0; nt < 6; ++nt) {
        const int n = n0 + nt * 16 + r;
        const float bn = bias[n];
        #pragma unroll
        for (int v = 0; v < 8; ++v) {
            long row = R0 + v + 8 * h8;
            long nb = row & 7, bt = row >> 3;
            long t = bt & (TT - 1), b = bt >> 8;
            size_t uxrow = (size_t)((b * NBLK + nb) * TT + t);
            UX[uxrow * NGATE + n] = acc[nt][v] + bn;
        }
    }
}

// ---------------------------------------------------------------------------
// Recurrent scan. Grid 32, block 512 (16 waves). Wave w owns hidden slice
// j in [16w,16w+16) for all three gates (W_hh B-frags resident in VGPRs).
__global__ __launch_bounds__(512) void gru_recur(
    const unsigned short* __restrict__ Whh,   // 768 x 256 bf16
    const float* __restrict__ UX,             // [seq][t][768] f32
    const float* __restrict__ bhh,            // 768
    unsigned short* __restrict__ Ybf,         // [seq][t][256] bf16 (or null)
    float* __restrict__ Yf32)                 // [b][t][nb][256] f32 (or null)
{
    const int tid  = threadIdx.x;
    const int wave = tid >> 5, lane = tid & 31;
    const int h8 = lane >> 4, r = lane & 15;
    const int j  = wave * 16 + r;             // hidden unit this lane produces
    const int seqbase = blockIdx.x * 16;

    __shared__ __align__(16) unsigned short hbuf[2][16 * 256]; // double-buffered h

    for (int i = tid; i < 2 * 16 * 256; i += 512) ((unsigned short*)hbuf)[i] = 0;

    // Persistent W_hh fragments: 3 gates x 8 K-tiles (192 VGPRs / lane).
    BFrag Bf[3][8];
    #pragma unroll
    for (int g = 0; g < 3; ++g) {
        const unsigned short* wp0 = Whh + (size_t)(g * CH + j) * CH + 8 * h8;
        #pragma unroll
        for (int kt = 0; kt < 8; ++kt) {
            Bf[g][kt].q[0] = *(const uint4n*)(wp0 + kt * 32);
            Bf[g][kt].q[1] = *(const uint4n*)(wp0 + kt * 32 + 16);
        }
    }
    const float bh0 = bhh[j], bh1 = bhh[CH + j], bh2 = bhh[2 * CH + j];
    __syncthreads();

    const v8f vzero = {0.f, 0.f, 0.f, 0.f, 0.f, 0.f, 0.f, 0.f};

    for (int t = 0; t < TT; ++t) {
        const unsigned short* hc = hbuf[t & 1];
        unsigned short*       hn = hbuf[(t & 1) ^ 1];

        v8f acc0 = vzero, acc1 = vzero, acc2 = vzero;
        #pragma unroll
        for (int kt = 0; kt < 8; ++kt) {
            BFrag af;
            const unsigned short* ha = hc + r * 256 + kt * 32 + 8 * h8;
            af.q[0] = *(const uint4n*)(ha);
            af.q[1] = *(const uint4n*)(ha + 16);
            acc0 = wmma_bf16(af, Bf[0][kt], acc0);
            acc1 = wmma_bf16(af, Bf[1][kt], acc1);
            acc2 = wmma_bf16(af, Bf[2][kt], acc2);
        }

        #pragma unroll
        for (int v = 0; v < 8; ++v) {
            const int m   = v + 8 * h8;
            const int seq = seqbase + m;
            const size_t uxoff = ((size_t)seq * TT + t) * NGATE + j;
            const float rx = UX[uxoff];
            const float zx = UX[uxoff + CH];
            const float nx = UX[uxoff + 2 * CH];
            if (t + 1 < TT) __builtin_prefetch(UX + uxoff + NGATE, 0, 0);

            const float rg = fsig(rx + acc0[v] + bh0);
            const float zg = fsig(zx + acc1[v] + bh1);
            const float ng = ftanh(nx + rg * (acc2[v] + bh2));
            const float hold = bf2f(hc[m * 256 + j]);
            const float hnew = (1.0f - zg) * ng + zg * hold;

            hn[m * 256 + j] = f2bf(hnew);
            if (Ybf)  Ybf[((size_t)seq * TT + t) * CH + j] = f2bf(hnew);
            if (Yf32) {
                const long b = seq >> 3, nb = seq & 7;
                Yf32[(((b * TT + t) * NBLK + nb) << 8) + j] = hnew;
            }
        }
        __syncthreads();
    }
}

// ---------------------------------------------------------------------------
extern "C" void kernel_launch(void* const* d_in, const int* in_sizes, int n_in,
                              void* d_out, int out_size, void* d_ws, size_t ws_size,
                              hipStream_t stream) {
    const float* x    = (const float*)d_in[0];
    const float* wih0 = (const float*)d_in[1];
    const float* whh0 = (const float*)d_in[2];
    const float* bih0 = (const float*)d_in[3];
    const float* bhh0 = (const float*)d_in[4];
    const float* wih1 = (const float*)d_in[5];
    const float* whh1 = (const float*)d_in[6];
    const float* bih1 = (const float*)d_in[7];
    const float* bhh1 = (const float*)d_in[8];
    float* out = (float*)d_out;

    // Workspace carving (~538 MB total).
    char* ws = (char*)d_ws;
    const size_t NX = (size_t)MROWS * 256;     // 33,554,432 elements
    const size_t NW = (size_t)NGATE * CH;      // 196,608 per weight matrix
    unsigned short* xbf   = (unsigned short*)ws; ws += NX * 2;
    unsigned short* wihb0 = (unsigned short*)ws; ws += NW * 2;
    unsigned short* whhb0 = (unsigned short*)ws; ws += NW * 2;
    unsigned short* wihb1 = (unsigned short*)ws; ws += NW * 2;
    unsigned short* whhb1 = (unsigned short*)ws; ws += NW * 2;
    float*          UX    = (float*)ws;          ws += (size_t)MROWS * NGATE * 4;
    unsigned short* ybf   = (unsigned short*)ws; ws += NX * 2;

    // Phase 0: f32 -> bf16 conversions.
    f32_to_bf16_kernel<<<4096, 256, 0, stream>>>(x, xbf, (long)NX);
    f32_to_bf16_kernel<<<192, 256, 0, stream>>>(wih0, wihb0, (long)NW);
    f32_to_bf16_kernel<<<192, 256, 0, stream>>>(whh0, whhb0, (long)NW);
    f32_to_bf16_kernel<<<192, 256, 0, stream>>>(wih1, wihb1, (long)NW);
    f32_to_bf16_kernel<<<192, 256, 0, stream>>>(whh1, whhb1, (long)NW);

    dim3 gemm_grid(MROWS / 128, NGATE / 96);   // (1024, 8)

    // Layer 0
    wmma_gemm_ux<<<gemm_grid, 256, 0, stream>>>(xbf, wihb0, bih0, UX, 0);
    gru_recur<<<NSEQ / 16, 512, 0, stream>>>(whhb0, UX, bhh0, ybf, nullptr);

    // Layer 1
    wmma_gemm_ux<<<gemm_grid, 256, 0, stream>>>(ybf, wihb1, bih1, UX, 1);
    gru_recur<<<NSEQ / 16, 512, 0, stream>>>(whhb1, UX, bhh1, nullptr, out);
}
